// PointNetSeg_8753143349324
// MI455X (gfx1250) — matrix operand order repaired and checked by reading, not compiled, not run on hardware
//
#include <hip/hip_runtime.h>

// ---------------------------------------------------------------------------
// PointNet segmentation forward for MI455X (gfx1250, wave32, WMMA).
//
// All big per-point GEMMs run through v_wmma_f32_16x16x32_f16 (f16 in, f32
// accumulate). Weight panels are staged into LDS once per block with
// GLOBAL_LOAD_ASYNC_TO_LDS_B128 (ASYNCcnt / s_wait_asynccnt) and consumed via
// ds_load_b128; activations stream from global. The inner loop is software-
// pipelined one k-step ahead so the ds_load -> wmma dependency is broken and
// the 4 WMMAs per step issue back-to-back. The three [N,1024] pre-pool
// tensors are never materialized: the WMMA epilogue tile-max-reduces and
// atomicMax'es into a [16,1024] pool (u32 atomic max is order-correct for
// non-negative post-ReLU floats).
// ---------------------------------------------------------------------------

typedef __attribute__((ext_vector_type(16))) _Float16 v16h;
typedef __attribute__((ext_vector_type(8)))  _Float16 v8h;
typedef __attribute__((ext_vector_type(8)))  float    v8f;
typedef __attribute__((ext_vector_type(4)))  int      v4i;

constexpr int BATCH = 16;
constexpr int PTS   = 4096;           // points per cloud
constexpr int NPT   = BATCH * PTS;    // 65536 points total
constexpr int LOG2P = 12;             // cloud id = point >> 12

#if defined(__has_builtin)
#  if __has_builtin(__builtin_amdgcn_global_load_async_to_lds_b128)
#    define HAVE_ASYNC_LDS 1
#  endif
#endif
#ifndef HAVE_ASYNC_LDS
#  define HAVE_ASYNC_LDS 0
#endif

// builtin signature: (int4 addrspace(1)*, int4 addrspace(3)*, imm off, imm cpol)
typedef __attribute__((address_space(1))) v4i* gv4_ptr;
typedef __attribute__((address_space(3))) v4i* lv4_ptr;

__device__ __forceinline__ void async_copy16(const _Float16* g, _Float16* l)
{
#if HAVE_ASYNC_LDS
    // flat LDS address truncated to 32 bits == LDS byte offset (aperture rule)
    __builtin_amdgcn_global_load_async_to_lds_b128(
        (gv4_ptr)(unsigned long long)g,
        (lv4_ptr)(unsigned)(unsigned long long)l, 0, 0);
#else
    *(v8h*)l = *(const v8h*)g;          // synchronous fallback
#endif
}

__device__ __forceinline__ void async_wait()
{
#if HAVE_ASYNC_LDS
#  if __has_builtin(__builtin_amdgcn_s_wait_asynccnt)
    __builtin_amdgcn_s_wait_asynccnt(0);
#  else
    asm volatile("s_wait_asynccnt 0x0" ::: "memory");
#  endif
#endif
}

__device__ __forceinline__ v16h join16(v8h lo, v8h hi)
{
    return __builtin_shufflevector(lo, hi, 0,1,2,3,4,5,6,7,8,9,10,11,12,13,14,15);
}

// ---------------------------------------------------------------------------
// WMMA GEMM:  out[M,K2] = act(A[M,K1] @ W[K1,K2] + bias)
//   A : f16 row-major, lda = K1 (K1 multiple of 32), streamed from global
//   WT: f16, W transposed -> [K2pad, K1]; the block's 16*NC-column panel is
//       async-staged into LDS (rows padded +8 halfs to spread banks)
//   NC: 16-col tiles per wave. grid = (M/128, K2/(16*NC)), 256 thr = 8 waves,
//       each wave owns a 16x(16*NC) strip; all 8 waves share the LDS panel.
//   POOL: skip store, tile-max + atomicMax into pool[cloud][col]
//   BW  : per-cloud weights (fstn 64x64 transform), stride K1*K2 per cloud
// ---------------------------------------------------------------------------
template<int NC, bool RELU, bool SF16, bool SF32, bool POOL, bool BW>
__global__ __launch_bounds__(256)
void wmma_gemm_kernel(const _Float16* __restrict__ A,
                      const _Float16* __restrict__ WT,
                      const float*    __restrict__ bias,
                      _Float16*       __restrict__ outH,
                      float*          __restrict__ outF,
                      unsigned*       __restrict__ pool,
                      int K1, int K2, int validCols)
{
    extern __shared__ _Float16 smem[];           // [16*NC][K1 + 8]
    const int ldsStride = K1 + 8;                // pad 16B to avoid bank camps

    const int lane    = threadIdx.x & 31;
    const int wave    = threadIdx.x >> 5;
    const int mrow    = lane & 15;               // M (A) / N (B,C) in tile
    const int half    = lane >> 4;               // K sub-block select
    const int rowBase = blockIdx.x * 128 + wave * 16;
    const int colBase = blockIdx.y * (16 * NC);
    const int cloud   = (blockIdx.x * 128) >> LOG2P;

    // ---- stage the 16*NC x K1 weight panel into LDS (async, once) ----
    {
        const _Float16* src = WT + (size_t)colBase * K1
                                 + (BW ? (size_t)cloud * K1 * K2 : 0);
        const int panel = 16 * NC * K1;          // always multiple of 2048
        for (int e = threadIdx.x * 8; e < panel; e += 256 * 8) {
            int col = e / K1;                    // 16B chunk stays in one col
            int w   = e - col * K1;
            async_copy16(src + e, smem + col * ldsStride + w);
        }
        async_wait();
        __syncthreads();
    }

    const _Float16* Arow = A + (size_t)(rowBase + mrow) * K1;
    const _Float16* Bbase[NC];
#pragma unroll
    for (int t = 0; t < NC; ++t)
        Bbase[t] = smem + (size_t)(t * 16 + mrow) * ldsStride;

    v8f acc[NC];
#pragma unroll
    for (int t = 0; t < NC; ++t) acc[t] = (v8f){};

    // ---- software-pipelined K loop: fetch k+32 while computing k ----
    // A fragment: lanes 0-15 K=k..k+15, lanes 16-31 K=k+16..k+31
    // B fragment from LDS: lanes 0-15 K=k..k+15, lanes 16-31 K=k+16..k+31
    v16h av = join16(*(const v8h*)(Arow + half * 16),
                     *(const v8h*)(Arow + half * 16 + 8));
    v16h bv[NC];
#pragma unroll
    for (int t = 0; t < NC; ++t)
        bv[t] = join16(*(const v8h*)(Bbase[t] + half * 16),
                       *(const v8h*)(Bbase[t] + half * 16 + 8));

    int k = 0;
    while (true) {
        const int kn = k + 32;
        v16h avn;
        v16h bvn[NC];
        if (kn < K1) {                            // uniform branch, EXEC intact
            avn = join16(*(const v8h*)(Arow + kn + half * 16),
                         *(const v8h*)(Arow + kn + half * 16 + 8));
#pragma unroll
            for (int t = 0; t < NC; ++t)
                bvn[t] = join16(*(const v8h*)(Bbase[t] + kn + half * 16),
                                *(const v8h*)(Bbase[t] + kn + half * 16 + 8));
        }
#pragma unroll
        for (int t = 0; t < NC; ++t) {
            acc[t] = __builtin_amdgcn_wmma_f32_16x16x32_f16(
                         false, av, false, bv[t], (short)0, acc[t], false, false);
        }
        if (kn >= K1) break;
        av = avn;
#pragma unroll
        for (int t = 0; t < NC; ++t) bv[t] = bvn[t];
        k = kn;
    }

#pragma unroll
    for (int t = 0; t < NC; ++t) {
        const int n = colBase + t * 16 + mrow;   // this lane's output column
        float bscal = 0.f;
        if (bias && n < validCols) bscal = bias[n];
#pragma unroll
        for (int r = 0; r < 8; ++r) {
            float v = acc[t][r] + bscal;
            if (RELU) v = fmaxf(v, 0.f);
            acc[t][r] = v;
        }
        if (SF16) {
#pragma unroll
            for (int r = 0; r < 8; ++r)          // C row = rowBase + r + 8*half
                outH[(size_t)(rowBase + half * 8 + r) * K2 + n] = (_Float16)acc[t][r];
        }
        if (SF32) {
            if (n < validCols) {
#pragma unroll
                for (int r = 0; r < 8; ++r)
                    outF[(size_t)(rowBase + half * 8 + r) * validCols + n] = acc[t][r];
            }
        }
        if (POOL) {
            float mx = acc[t][0];
#pragma unroll
            for (int r = 1; r < 8; ++r) mx = fmaxf(mx, acc[t][r]);
            mx = fmaxf(mx, __shfl_xor(mx, 16, 32));  // combine M halves
            if (half == 0)                           // one atomic per column
                atomicMax(pool + (size_t)cloud * K2 + n, __float_as_uint(mx));
        }
    }
}

// ------------------------- support kernels ---------------------------------

// Convert W[K1,K2] f32 -> WT[K2p,K1p] f16 (transposed, zero padded)
__global__ void conv_wT_kernel(const float* __restrict__ W, _Float16* __restrict__ WT,
                               int K1, int K2, int K1p, int K2p)
{
    int idx = blockIdx.x * 256 + threadIdx.x;
    if (idx >= K1p * K2p) return;
    int c = idx / K1p, r = idx - c * K1p;          // WT[c][r] = W[r][c]
    float v = (r < K1 && c < K2) ? W[(size_t)r * K2 + c] : 0.f;
    WT[idx] = (_Float16)v;
}

__global__ void zero_u32_kernel(unsigned* __restrict__ p, int n)
{
    int i = blockIdx.x * 256 + threadIdx.x;
    if (i < n) p[i] = 0u;
}

// x[N,3] f32 -> (optional 3x3 STN transform) -> xp[N,32] f16 zero-padded
__global__ void pack_x_kernel(const float* __restrict__ x,
                              const float* __restrict__ T3,  // [16,9] or null
                              _Float16* __restrict__ xp)
{
    int i = blockIdx.x * 256 + threadIdx.x;        // exact grid, no guard
    float v0 = x[3 * i + 0], v1 = x[3 * i + 1], v2 = x[3 * i + 2];
    float o0 = v0, o1 = v1, o2 = v2;
    if (T3) {
        const float* T = T3 + (i >> LOG2P) * 9;
        o0 = v0 * T[0] + v1 * T[3] + v2 * T[6];
        o1 = v0 * T[1] + v1 * T[4] + v2 * T[7];
        o2 = v0 * T[2] + v1 * T[5] + v2 * T[8];
    }
    _Float16* row = xp + (size_t)i * 32;
    row[0] = (_Float16)o0; row[1] = (_Float16)o1; row[2] = (_Float16)o2;
#pragma unroll
    for (int j = 3; j < 32; ++j) row[j] = (_Float16)0.f;
}

// tiny [16,K1] @ [K1,K2] + b, ReLU, f32 (global MLP after pooling)
__global__ void small_mlp_kernel(const float* __restrict__ in, const float* __restrict__ W,
                                 const float* __restrict__ b, float* __restrict__ out,
                                 int K1, int K2)
{
    int idx = blockIdx.x * 256 + threadIdx.x;
    if (idx >= BATCH * K2) return;
    int row = idx / K2, col = idx - row * K2;
    float s = b[col];
    const float* ir = in + (size_t)row * K1;
    for (int k = 0; k < K1; ++k) s = fmaf(ir[k], W[(size_t)k * K2 + col], s);
    out[idx] = fmaxf(s, 0.f);
}

// istn fc: t3[16,9] = g[16,256] @ W[256,9] + b + I3 (f32, fed to pack_x)
__global__ void trans3_kernel(const float* __restrict__ g, const float* __restrict__ W,
                              const float* __restrict__ b, float* __restrict__ t3)
{
    int idx = blockIdx.x * 256 + threadIdx.x;
    if (idx >= BATCH * 9) return;
    int row = idx / 9, j = idx - row * 9;
    float s = b[j] + ((j == 0 || j == 4 || j == 8) ? 1.f : 0.f);
    const float* gr = g + row * 256;
    for (int k = 0; k < 256; ++k) s = fmaf(gr[k], W[(size_t)k * 9 + j], s);
    t3[idx] = s;
}

// fstn fc: per-cloud 64x64 transform (+I), stored f16 TRANSPOSED for WMMA-B
__global__ void trans64T_kernel(const float* __restrict__ g, const float* __restrict__ W,
                                const float* __restrict__ b, _Float16* __restrict__ wt)
{
    int idx = blockIdx.x * 256 + threadIdx.x;      // 16*4096 exact
    int bb = idx >> 12, j = idx & 4095;
    int r = j >> 6, c = j & 63;                    // trans[bb][r][c]
    float s = b[j] + ((r == c) ? 1.f : 0.f);
    const float* gr = g + bb * 256;
    for (int k = 0; k < 256; ++k) s = fmaf(gr[k], W[(size_t)k * 4096 + j], s);
    wt[(size_t)bb * 4096 + c * 64 + r] = (_Float16)s;  // WT[c][r]
}

// feat[N,1088] f16 = concat(xf[N,64], pool[cloud][1024])
__global__ void build_feat_kernel(const _Float16* __restrict__ xf,
                                  const unsigned* __restrict__ pool,
                                  _Float16* __restrict__ feat)
{
    int idx = blockIdx.x * 256 + threadIdx.x;      // N*1088 exact
    int i = idx / 1088, c = idx - i * 1088;
    _Float16 v;
    if (c < 64) v = xf[(size_t)i * 64 + c];
    else        v = (_Float16)__uint_as_float(pool[(size_t)(i >> LOG2P) * 1024 + (c - 64)]);
    feat[idx] = v;
}

// ---------------------------------------------------------------------------
extern "C" void kernel_launch(void* const* d_in, const int* in_sizes, int n_in,
                              void* d_out, int out_size, void* d_ws, size_t ws_size,
                              hipStream_t stream)
{
    (void)in_sizes; (void)n_in; (void)out_size; (void)ws_size;

    const float* x = (const float*)d_in[0];
    // d_in[1] = batch ids (unused: clouds are equal-size & sorted -> i>>12)
    const float* iW0  = (const float*)d_in[2];  const float* ib0  = (const float*)d_in[3];
    const float* iW1  = (const float*)d_in[4];  const float* ib1  = (const float*)d_in[5];
    const float* iW2  = (const float*)d_in[6];  const float* ib2  = (const float*)d_in[7];
    const float* igW0 = (const float*)d_in[8];  const float* igb0 = (const float*)d_in[9];
    const float* igW1 = (const float*)d_in[10]; const float* igb1 = (const float*)d_in[11];
    const float* ifcW = (const float*)d_in[12]; const float* ifcb = (const float*)d_in[13];
    const float* l1W0 = (const float*)d_in[14]; const float* l1b0 = (const float*)d_in[15];
    const float* l1W1 = (const float*)d_in[16]; const float* l1b1 = (const float*)d_in[17];
    const float* fW0  = (const float*)d_in[18]; const float* fb0  = (const float*)d_in[19];
    const float* fW1  = (const float*)d_in[20]; const float* fb1  = (const float*)d_in[21];
    const float* fW2  = (const float*)d_in[22]; const float* fb2  = (const float*)d_in[23];
    const float* fgW0 = (const float*)d_in[24]; const float* fgb0 = (const float*)d_in[25];
    const float* fgW1 = (const float*)d_in[26]; const float* fgb1 = (const float*)d_in[27];
    const float* ffcW = (const float*)d_in[28]; const float* ffcb = (const float*)d_in[29];
    const float* l2W0 = (const float*)d_in[30]; const float* l2b0 = (const float*)d_in[31];
    const float* l2W1 = (const float*)d_in[32]; const float* l2b1 = (const float*)d_in[33];
    const float* l2W2 = (const float*)d_in[34]; const float* l2b2 = (const float*)d_in[35];
    const float* sW0  = (const float*)d_in[36]; const float* sb0  = (const float*)d_in[37];
    const float* sW1  = (const float*)d_in[38]; const float* sb1  = (const float*)d_in[39];
    const float* sW2  = (const float*)d_in[40]; const float* sb2  = (const float*)d_in[41];
    const float* sW3  = (const float*)d_in[42]; const float* sb3  = (const float*)d_in[43];

    // ----- workspace bump allocator -----
    unsigned char* base = (unsigned char*)d_ws;
    size_t off = 0;
    auto alloc = [&](size_t bytes) -> void* {
        void* p = base + off;
        off = (off + bytes + 255) & ~(size_t)255;
        return p;
    };
    auto allocH = [&](size_t elems) { return (_Float16*)alloc(elems * 2); };

    // transposed f16 weights
    _Float16* wt_i0  = allocH(64 * 32);
    _Float16* wt_i1  = allocH(128 * 64);
    _Float16* wt_i2  = allocH(1024 * 128);
    _Float16* wt_l10 = allocH(64 * 32);
    _Float16* wt_l11 = allocH(64 * 64);
    _Float16* wt_f0  = allocH(64 * 64);
    _Float16* wt_f1  = allocH(128 * 64);
    _Float16* wt_f2  = allocH(1024 * 128);
    _Float16* wt_l20 = allocH(64 * 64);
    _Float16* wt_l21 = allocH(128 * 64);
    _Float16* wt_l22 = allocH(1024 * 128);
    _Float16* wt_s0  = allocH(512 * 1088);
    _Float16* wt_s1  = allocH(256 * 512);
    _Float16* wt_s2  = allocH(128 * 256);
    _Float16* wt_s3  = allocH(16 * 128);
    _Float16* wt64   = allocH(BATCH * 64 * 64);      // per-cloud fstn transforms

    unsigned* pool = (unsigned*)alloc(BATCH * 1024 * 4);
    float*    g512 = (float*)alloc(BATCH * 512 * 4);
    float*    g256 = (float*)alloc(BATCH * 256 * 4);
    float*    t3   = (float*)alloc(BATCH * 9 * 4);

    _Float16* XP   = allocH((size_t)NPT * 32);
    _Float16* H64  = allocH((size_t)NPT * 64);
    _Float16* A64  = allocH((size_t)NPT * 64);       // local1 output (kept)
    _Float16* XF   = allocH((size_t)NPT * 64);       // transformed features (kept)
    _Float16* H128 = allocH((size_t)NPT * 128);
    _Float16* H512 = allocH((size_t)NPT * 512);
    _Float16* H256 = allocH((size_t)NPT * 256);
    _Float16* FEAT = allocH((size_t)NPT * 1088);

    auto convW = [&](const float* W, _Float16* WT, int K1, int K2, int K1p, int K2p) {
        int total = K1p * K2p;
        conv_wT_kernel<<<(total + 255) / 256, 256, 0, stream>>>(W, WT, K1, K2, K1p, K2p);
    };
    auto zeroPool = [&]() {
        zero_u32_kernel<<<(BATCH * 1024) / 256, 256, 0, stream>>>(pool, BATCH * 1024);
    };

    const dim3 blk(256);

    // dynamic LDS: 16*NC columns x (K1 + 8) halfs
#define SMEM_BYTES(NC_, K1_) ((size_t)(16 * (NC_)) * ((K1_) + 8) * 2)
#define GEMM4(A_, WT_, B_, OH_, K1_, K2_) \
    wmma_gemm_kernel<4, true, true, false, false, false><<<dim3(NPT/128, (K2_)/64), blk, SMEM_BYTES(4, K1_), stream>>>( \
        A_, WT_, B_, OH_, nullptr, nullptr, K1_, K2_, K2_)
#define GEMM4_POOL(A_, WT_, B_, K1_, K2_) \
    wmma_gemm_kernel<4, true, false, false, true, false><<<dim3(NPT/128, (K2_)/64), blk, SMEM_BYTES(4, K1_), stream>>>( \
        A_, WT_, B_, nullptr, nullptr, pool, K1_, K2_, K2_)

    // ----- weight prep (f16 transposed, padded) -----
    convW(iW0,  wt_i0,  3, 64, 32, 64);
    convW(iW1,  wt_i1,  64, 128, 64, 128);
    convW(iW2,  wt_i2,  128, 1024, 128, 1024);
    convW(l1W0, wt_l10, 3, 64, 32, 64);
    convW(l1W1, wt_l11, 64, 64, 64, 64);
    convW(fW0,  wt_f0,  64, 64, 64, 64);
    convW(fW1,  wt_f1,  64, 128, 64, 128);
    convW(fW2,  wt_f2,  128, 1024, 128, 1024);
    convW(l2W0, wt_l20, 64, 64, 64, 64);
    convW(l2W1, wt_l21, 64, 128, 64, 128);
    convW(l2W2, wt_l22, 128, 1024, 128, 1024);
    convW(sW0,  wt_s0,  1088, 512, 1088, 512);
    convW(sW1,  wt_s1,  512, 256, 512, 256);
    convW(sW2,  wt_s2,  256, 128, 256, 128);
    convW(sW3,  wt_s3,  128, 4, 128, 16);

    // ----- input STN (3x3) -----
    pack_x_kernel<<<NPT / 256, 256, 0, stream>>>(x, nullptr, XP);
    GEMM4(XP,  wt_i0, ib0, H64,  32, 64);
    GEMM4(H64, wt_i1, ib1, H128, 64, 128);
    zeroPool();
    GEMM4_POOL(H128, wt_i2, ib2, 128, 1024);         // pooled, never stored
    small_mlp_kernel<<<(BATCH*512 + 255)/256, 256, 0, stream>>>((const float*)pool, igW0, igb0, g512, 1024, 512);
    small_mlp_kernel<<<(BATCH*256 + 255)/256, 256, 0, stream>>>(g512, igW1, igb1, g256, 512, 256);
    trans3_kernel<<<1, 256, 0, stream>>>(g256, ifcW, ifcb, t3);
    pack_x_kernel<<<NPT / 256, 256, 0, stream>>>(x, t3, XP);   // apply 3x3 + repack

    // ----- local1 -----
    GEMM4(XP,  wt_l10, l1b0, H64, 32, 64);
    GEMM4(H64, wt_l11, l1b1, A64, 64, 64);

    // ----- feature STN (64x64) -----
    GEMM4(A64, wt_f0, fb0, H64,  64, 64);
    GEMM4(H64, wt_f1, fb1, H128, 64, 128);
    zeroPool();
    GEMM4_POOL(H128, wt_f2, fb2, 128, 1024);
    small_mlp_kernel<<<(BATCH*512 + 255)/256, 256, 0, stream>>>((const float*)pool, fgW0, fgb0, g512, 1024, 512);
    small_mlp_kernel<<<(BATCH*256 + 255)/256, 256, 0, stream>>>(g512, fgW1, fgb1, g256, 512, 256);
    trans64T_kernel<<<(BATCH*4096) / 256, 256, 0, stream>>>(g256, ffcW, ffcb, wt64);
    // XF = A64 @ T_b  (per-cloud weights, no bias, no relu)
    wmma_gemm_kernel<4, false, true, false, false, true><<<dim3(NPT/128, 1), blk, SMEM_BYTES(4, 64), stream>>>(
        A64, wt64, nullptr, XF, nullptr, nullptr, 64, 64, 64);

    // ----- local2 (pooled only) -----
    GEMM4(XF,  wt_l20, l2b0, H64,  64, 64);
    GEMM4(H64, wt_l21, l2b1, H128, 64, 128);
    zeroPool();
    GEMM4_POOL(H128, wt_l22, l2b2, 128, 1024);

    // ----- concat [xf | gf[batch]] -----
    build_feat_kernel<<<((size_t)NPT * 1088) / 256, 256, 0, stream>>>(XF, pool, FEAT);

    // ----- segmentation head -----
    GEMM4(FEAT, wt_s0, sb0, H512, 1088, 512);
    GEMM4(H512, wt_s1, sb1, H256, 512, 256);
    GEMM4(H256, wt_s2, sb2, H128, 256, 128);
    // final 128 -> 4 (padded to 16 cols, f32 store with col guard, ReLU)
    wmma_gemm_kernel<1, true, false, true, false, false><<<dim3(NPT/128, 1), blk, SMEM_BYTES(1, 128), stream>>>(
        H128, wt_s3, sb3, nullptr, (float*)d_out, nullptr, 128, 16, 4);

#undef GEMM4
#undef GEMM4_POOL
#undef SMEM_BYTES
}
